// TemporalExtGCN_71657234366602
// MI455X (gfx1250) — matrix operand order repaired and checked by reading, not compiled, not run on hardware
//
#include <hip/hip_runtime.h>

typedef __attribute__((ext_vector_type(16))) __bf16 v16bf;
typedef __attribute__((ext_vector_type(8)))  float  v8f;

#define N_NODES  100000
#define N_EDGES  1600000
#define IN_DIM   500
#define KPAD     512
#define HID      32
#define N_REL    7
#define OUT_DIM  4
#define N_GRAPHS 16
#define COLS     256      /* 32 root + 7*32 relation cols */
#define RELCOLS  224
#define NKB      (KPAD/32)   /* 16 k-blocks */
#define NNB      (COLS/16)   /* 16 n-blocks */
#define NSTRIPS  (N_NODES/16) /* 6250 */

// ---------------------------------------------------------------------------
// Pre-swizzle W_root|W_rel (fp32) into WMMA B-fragment order (bf16).
// B-frag layout (16x16x32 bf16, ISA 7.12.2): lane<16: N=lane, elem i<8 -> K=kb*32+i, i>=8 -> K=kb*32+8+i
//                                            lane>=16: N=lane-16, elem i<8 -> K=kb*32+8+i, i>=8 -> K=kb*32+16+i
// Stored as [kb][nb][lane] x v16bf so each lane does one 32B load in the GEMM.
// ---------------------------------------------------------------------------
__global__ void k_prep_w(const float* __restrict__ W_rel,
                         const float* __restrict__ W_root,
                         v16bf* __restrict__ bfrag) {
    int t = blockIdx.x * blockDim.x + threadIdx.x;   // 0 .. 16*16*32-1
    if (t >= NKB * NNB * 32) return;
    int lane = t & 31;
    int nb   = (t >> 5) & 15;
    int kb   = t >> 9;
    int c    = nb * 16 + (lane & 15);
    int kofs = (lane < 16) ? 0 : 8;
    v16bf v;
#pragma unroll
    for (int i = 0; i < 16; ++i) {
        int K = kb * 32 + kofs + (i < 8 ? i : 8 + i);
        float w = 0.0f;
        if (K < IN_DIM) {
            if (c < HID) w = W_root[K * HID + c];
            else {
                int r = (c >> 5) - 1;
                w = W_rel[((size_t)r * IN_DIM + K) * HID + (c & 31)];
            }
        }
        v[i] = (__bf16)w;
    }
    bfrag[t] = v;
}

__global__ void k_zero_u32(unsigned* __restrict__ p, int n) {
    int t = blockIdx.x * blockDim.x + threadIdx.x;
    if (t < n) p[t] = 0u;
}

// per-(dst, relation) in-degree counts
__global__ void k_count(const int* __restrict__ edge_index,
                        const int* __restrict__ edge_attr,
                        int* __restrict__ cnt) {
    int e = blockIdx.x * blockDim.x + threadIdx.x;
    if (e >= N_EDGES) return;
    int r   = edge_attr[e];
    int dst = edge_index[N_EDGES + e];
    atomicAdd(&cnt[dst * 8 + r], 1);
}

// ---------------------------------------------------------------------------
// Fused GEMM: H256 = bf16(node_x) @ Wb   via v_wmma_f32_16x16x32_bf16.
// One wave per 16-row strip; 16 fp32 accumulators cover all 256 cols.
// A (node_x rows) double-buffered as aligned float4 b128 loads; B fragments
// double-buffered so each WMMA overlaps the next fragment's load.
// Cols 0..31 (+bias) -> out_acc (root transform), cols 32..255 -> H (messages).
// ---------------------------------------------------------------------------
__global__ void __launch_bounds__(128)
k_gemm(const float* __restrict__ node_x,
       const v16bf* __restrict__ bfrag,
       const float* __restrict__ bias,
       float* __restrict__ out_acc,
       float* __restrict__ H) {
    int strip = blockIdx.x * 4 + (threadIdx.x >> 5);
    if (strip >= NSTRIPS) return;
    int lane = threadIdx.x & 31;
    int row0 = strip * 16;

    // Row stride 500 floats = 2000 B (multiple of 16) and K offsets are
    // multiples of 8 floats => all float4 accesses below are 16B aligned.
    const float* __restrict__ xr = node_x + (size_t)(row0 + (lane & 15)) * IN_DIM;
    int kbase = (lane < 16) ? 0 : 8;   // ISA A-frag K base for this half-wave

    v8f acc[NNB];
#pragma unroll
    for (int nb = 0; nb < NNB; ++nb) acc[nb] = (v8f){0,0,0,0,0,0,0,0};

    // Convert 16 raw fp32 A values (two aligned groups of 8) to a bf16 A-frag.
    auto cvtA = [](float4 a0, float4 a1, float4 a2, float4 a3) -> v16bf {
        v16bf a;
        a[0]=(__bf16)a0.x; a[1]=(__bf16)a0.y; a[2]=(__bf16)a0.z; a[3]=(__bf16)a0.w;
        a[4]=(__bf16)a1.x; a[5]=(__bf16)a1.y; a[6]=(__bf16)a1.z; a[7]=(__bf16)a1.w;
        a[8]=(__bf16)a2.x; a[9]=(__bf16)a2.y; a[10]=(__bf16)a2.z; a[11]=(__bf16)a2.w;
        a[12]=(__bf16)a3.x; a[13]=(__bf16)a3.y; a[14]=(__bf16)a3.z; a[15]=(__bf16)a3.w;
        return a;
    };

    // B fragment pipeline register (double-buffered inside mm()).
    v16bf bcur = bfrag[lane];   // kb=0, nb=0

    // One K-block of 16 WMMAs with B prefetch one fragment ahead.
    auto mm = [&](v16bf a, int kb, bool lastkb) {
        const v16bf* __restrict__ bp = bfrag + (size_t)kb * NNB * 32 + lane;
#pragma unroll
        for (int nb = 0; nb < NNB; ++nb) {
            v16bf bn = bcur;
            if (nb < NNB - 1)  bn = bp[(size_t)(nb + 1) * 32];
            else if (!lastkb)  bn = bp[(size_t)NNB * 32];   // first frag of next kb
            acc[nb] = __builtin_amdgcn_wmma_f32_16x16x32_bf16(
                false, a, false, bn /*placeholder*/, (short)0, acc[nb], false, false);
            // NOTE: use bcur for the multiply, bn only flows forward:
            acc[nb] = acc[nb]; // (see corrected call below)
            bcur = bn;
        }
    };
    (void)mm; // replaced by mm2 below (kept simple & correct)

    auto mm2 = [&](v16bf a, int kb, bool lastkb) {
        const v16bf* __restrict__ bp = bfrag + (size_t)kb * NNB * 32 + lane;
#pragma unroll
        for (int nb = 0; nb < NNB; ++nb) {
            v16bf bn = bcur;
            if (nb < NNB - 1)  bn = bp[(size_t)(nb + 1) * 32];
            else if (!lastkb)  bn = bp[(size_t)NNB * 32];
            acc[nb] = __builtin_amdgcn_wmma_f32_16x16x32_bf16(
                false, a, false, bcur, (short)0, acc[nb], false, false);
            bcur = bn;
        }
    };

    // Preload A for kb=0 (all K < 500).
    const float4* p0 = (const float4*)(xr + kbase);
    float4 q0 = p0[0], q1 = p0[1], q2 = p0[4], q3 = p0[5];

    // kb = 0..13: uniform, branch-free, full b128 prefetch of kb+1.
    for (int kb = 0; kb < 14; ++kb) {
        v16bf a = cvtA(q0, q1, q2, q3);
        const float4* np = (const float4*)(xr + (kb + 1) * 32 + kbase);
        float4 n0 = np[0], n1 = np[1], n2 = np[4], n3 = np[5];
        mm2(a, kb, false);
        q0 = n0; q1 = n1; q2 = n2; q3 = n3;
    }

    // kb = 14: prefetch masked tail block (K = 480..511, valid K < 500).
    {
        v16bf a = cvtA(q0, q1, q2, q3);
        const float4* tp = (const float4*)(xr + 15 * 32 + kbase); // K=480 or 488
        float4 n0 = tp[0], n1 = tp[1];                 // K .. K+7 all < 500
        float4 n2 = make_float4(0.f, 0.f, 0.f, 0.f);
        if (lane < 16) n2 = *(const float4*)(xr + 496); // K=496..499 valid
        float4 n3 = make_float4(0.f, 0.f, 0.f, 0.f);    // K>=500 -> 0
        mm2(a, 14, false);
        q0 = n0; q1 = n1; q2 = n2; q3 = n3;
    }

    // kb = 15 (tail).
    {
        v16bf a = cvtA(q0, q1, q2, q3);
        mm2(a, 15, true);
    }

    // Epilogue: C/D layout — lane<16: N=lane, M=g; lane>=16: N=lane-16, M=8+g
    int cbase = lane & 15;
    int mofs  = (lane < 16) ? 0 : 8;
#pragma unroll
    for (int nb = 0; nb < NNB; ++nb) {
        int c = nb * 16 + cbase;
#pragma unroll
        for (int g = 0; g < 8; ++g) {
            int row = row0 + mofs + g;
            float v = acc[nb][g];
            if (c < HID) out_acc[(size_t)row * HID + c] = v + bias[c];
            else         H[(size_t)row * RELCOLS + (c - HID)] = v;
        }
    }
}

// scatter: out_acc[dst] += H[src, rel-block] / cnt[dst,rel]   (32 lanes = 32 cols)
__global__ void k_scatter(const int* __restrict__ edge_index,
                          const int* __restrict__ edge_attr,
                          const int* __restrict__ cnt,
                          const float* __restrict__ H,
                          float* __restrict__ out_acc) {
    int t = blockIdx.x * blockDim.x + threadIdx.x;
    int e = t >> 5;
    if (e >= N_EDGES) return;
    int j   = t & 31;
    int r   = edge_attr[e];
    int src = edge_index[e];
    int dst = edge_index[N_EDGES + e];
    int c   = cnt[dst * 8 + r];
    float inv = 1.0f / (float)(c > 1 ? c : 1);
    float v = H[(size_t)src * RELCOLS + r * HID + j] * inv;
    unsafeAtomicAdd(&out_acc[(size_t)dst * HID + j], v);
}

// relu + global_add_pool: LDS partials per block (batch sorted => low contention)
__global__ void k_pool(const float* __restrict__ out_acc,
                       const int* __restrict__ batch,
                       float* __restrict__ pooled) {
    __shared__ float part[N_GRAPHS * HID];
    for (int i = threadIdx.x; i < N_GRAPHS * HID; i += 256) part[i] = 0.0f;
    __syncthreads();
    int j   = threadIdx.x & 31;
    int sub = threadIdx.x >> 5;            // 0..7
    int base = blockIdx.x * 128;
#pragma unroll
    for (int it = 0; it < 16; ++it) {
        int n = base + it * 8 + sub;
        if (n < N_NODES) {
            int g = batch[n];
            float v = out_acc[(size_t)n * HID + j];
            v = v > 0.0f ? v : 0.0f;
            atomicAdd(&part[g * HID + j], v);  // ds_add_f32
        }
    }
    __syncthreads();
    for (int i = threadIdx.x; i < N_GRAPHS * HID; i += 256) {
        float v = part[i];
        if (v != 0.0f) unsafeAtomicAdd(&pooled[i], v);
    }
}

__global__ void k_fc(const float* __restrict__ pooled,
                     const float* __restrict__ fc_w,
                     const float* __restrict__ fc_b,
                     float* __restrict__ out) {
    int t = threadIdx.x;
    if (t >= N_GRAPHS * OUT_DIM) return;
    int g = t >> 2, o = t & 3;
    float s = fc_b[o];
#pragma unroll
    for (int jj = 0; jj < HID; ++jj) s += pooled[g * HID + jj] * fc_w[jj * OUT_DIM + o];
    out[g * OUT_DIM + o] = s;
}

extern "C" void kernel_launch(void* const* d_in, const int* in_sizes, int n_in,
                              void* d_out, int out_size, void* d_ws, size_t ws_size,
                              hipStream_t stream) {
    const float* node_x     = (const float*)d_in[0];
    const int*   edge_index = (const int*)d_in[1];
    const int*   edge_attr  = (const int*)d_in[2];
    const int*   batch      = (const int*)d_in[3];
    const float* W_rel      = (const float*)d_in[4];
    const float* W_root     = (const float*)d_in[5];
    const float* bias       = (const float*)d_in[6];
    const float* fc_w       = (const float*)d_in[7];
    const float* fc_b       = (const float*)d_in[8];
    float* out = (float*)d_out;

    // workspace layout (256B aligned)
    char* ws = (char*)d_ws;
    size_t off = 0;
    v16bf* bfrag = (v16bf*)(ws + off);            off += (size_t)NKB * NNB * 32 * sizeof(v16bf); // 256 KB
    int*   cnt   = (int*)(ws + off);              off += (size_t)N_NODES * 8 * sizeof(int);      // 3.2 MB
    float* out_acc = (float*)(ws + off);          off += (size_t)N_NODES * HID * sizeof(float);  // 12.8 MB
    float* H     = (float*)(ws + off);            off += (size_t)N_NODES * RELCOLS * sizeof(float); // 89.6 MB
    float* pooled = (float*)(ws + off);           off += (size_t)N_GRAPHS * HID * sizeof(float);
    (void)ws_size; (void)in_sizes; (void)n_in; (void)out_size;

    k_prep_w<<<(NKB * NNB * 32 + 255) / 256, 256, 0, stream>>>(W_rel, W_root, bfrag);
    k_zero_u32<<<(N_NODES * 8 + 255) / 256, 256, 0, stream>>>((unsigned*)cnt, N_NODES * 8);
    k_zero_u32<<<(N_GRAPHS * HID + 255) / 256, 256, 0, stream>>>((unsigned*)pooled, N_GRAPHS * HID);
    k_count<<<(N_EDGES + 255) / 256, 256, 0, stream>>>(edge_index, edge_attr, cnt);
    k_gemm<<<(NSTRIPS + 3) / 4, 128, 0, stream>>>(node_x, bfrag, bias, out_acc, H);
    k_scatter<<<((size_t)N_EDGES * 32 + 255) / 256, 256, 0, stream>>>(edge_index, edge_attr, cnt, H, out_acc);
    k_pool<<<(N_NODES + 127) / 128, 256, 0, stream>>>(out_acc, batch, pooled);
    k_fc<<<1, 64, 0, stream>>>(pooled, fc_w, fc_b, out);
}
